// GAT_gate_11501922419054
// MI455X (gfx1250) — matrix operand order repaired and checked by reading, not compiled, not run on hardware
//
#include <hip/hip_runtime.h>

// ---------------------------------------------------------------------------
// GAT + gate, MI455X (gfx1250, wave32, WMMA f16 16x16x32 -> f32 acc,
// TDM async tensor loads for the flash-attention column tiles)
// N=8192 nodes, D=256 feat, E=131072 edges
// ---------------------------------------------------------------------------

#define NN 8192
#define DD 256
#define EE 131072
#define BN 32            // column tile of the flash loop (== 32 bits -> 1 adj word)

typedef __attribute__((ext_vector_type(16))) _Float16 v16h;
typedef __attribute__((ext_vector_type(8)))  _Float16 v8h;
typedef __attribute__((ext_vector_type(8)))  float    v8f;
typedef __attribute__((ext_vector_type(4)))  unsigned int v4u;
typedef __attribute__((ext_vector_type(8)))  int      v8i;
typedef __attribute__((ext_vector_type(4)))  int      v4i;

__device__ __forceinline__ v8f wmma16(v16h a, v16h b, v8f c) {
  // D = A(16x32 f16) * B(32x16 f16) + C(16x16 f32)
  return __builtin_amdgcn_wmma_f32_16x16x32_f16(
      /*neg_a=*/false, a, /*neg_b=*/false, b,
      /*c_mod=*/(short)0, c, /*reuse_a=*/false, /*reuse_b=*/false);
}

// ---- Tensor Data Mover: 1-D contiguous copy global -> LDS -----------------
// D# per CDNA5 ISA: count=1, type=2 ("image"),
// data_size=3 (8B), tile_dim0 = tensor_dim0 = tensor_dim0_stride = qwords,
// tensor_dim1 = tile_dim1 = 1, groups 2/3 zero (<=2D tensor).
__device__ __forceinline__ void tdm_load_1d(const void* gsrc, unsigned lds_off,
                                            unsigned qwords) {
  unsigned long long ga = (unsigned long long)(size_t)gsrc;
  v4u g0 = (v4u)0u;
  g0[0] = 1u;                                              // count=1, user mode
  g0[1] = lds_off;                                         // lds_addr (bytes)
  g0[2] = (unsigned)(ga & 0xffffffffull);                  // global_addr[31:0]
  g0[3] = (unsigned)((ga >> 32) & 0x01ffffffull)           // global_addr[56:32]
        | (2u << 30);                                      // type = 2
  v8i g1 = (v8i)0;
  g1[0] = (int)(3u << 16);                                 // data_size = 8B
  g1[1] = (int)(qwords << 16);                             // tensor_dim0[15:0] @63:48
  g1[2] = (int)(1u << 16);                                 // tensor_dim1 = 1 @111:80
  g1[3] = (int)(qwords << 16);                             // tile_dim0 @127:112
  g1[4] = 1;                                               // tile_dim1 = 1 @143:128
  g1[5] = (int)qwords;                                     // tensor_dim0_stride
  v4i gz4 = (v4i)0;
#if __clang_major__ >= 23
  v8i gz8 = (v8i)0;
  __builtin_amdgcn_tensor_load_to_lds(g0, g1, gz4, gz4, gz8, 0);
#else
  __builtin_amdgcn_tensor_load_to_lds(g0, g1, gz4, gz4, 0);
#endif
}

// A fragment (16x32, M x K), row-major f16 source. Contiguous k order (k = 16h + s, as in load_b_h), not the ISA table order:
// lanes 0-15: elems 0..15 = K k..k+15 (row = lane)
// lanes 16-31: same rows, K offset +16.
__device__ __forceinline__ v16h load_a_h(const _Float16* __restrict__ base,
                                         int ld, int row, int k, int sel) {
  const _Float16* p = base + row * ld + k + sel * 16;
  v8h lo = *(const v8h*)(p);
  v8h hi = *(const v8h*)(p + 8);
  v16h r;
#pragma unroll
  for (int i = 0; i < 8; ++i) { r[i] = lo[i]; r[8 + i] = hi[i]; }
  return r;
}

__device__ __forceinline__ v16h load_a_f32(const float* __restrict__ base,
                                           int ld, int row, int k, int sel) {
  const float* p = base + row * ld + k + sel * 16;
  v16h r;
#pragma unroll
  for (int i = 0; i < 8; ++i) {
    r[i]     = (_Float16)p[i];
    r[8 + i] = (_Float16)p[8 + i];
  }
  return r;
}

// B fragment (32x16, K x N) from a [N][K] row-major source (i.e. B^T row-major):
// lane = column N (mod 16); lanes 0-15 take K k..k+15, lanes 16-31 take k+16..k+31.
__device__ __forceinline__ v16h load_b_h(const _Float16* __restrict__ base,
                                         int ld, int col, int k, int sel) {
  return *(const v16h*)(base + col * ld + k + sel * 16);
}

__device__ __forceinline__ v16h load_b_f32(const float* __restrict__ base,
                                           int ld, int col, int k, int sel) {
  const float* p = base + col * ld + k + sel * 16;
  v16h r;
#pragma unroll
  for (int i = 0; i < 16; ++i) r[i] = (_Float16)p[i];
  return r;
}

// ---------------------------------------------------------------------------
// Kernel 0: zero the adjacency bitmask (8 MB)
// ---------------------------------------------------------------------------
__global__ void k_zero(uint4* __restrict__ p) {
  int i = blockIdx.x * blockDim.x + threadIdx.x;
  p[i] = make_uint4(0u, 0u, 0u, 0u);
}

// ---------------------------------------------------------------------------
// Kernel 1: adjacency bitmask from edge_index (int64) + self loops
// adj bit layout: word = adj[row*256 + col/32], bit = col%32
// ---------------------------------------------------------------------------
__global__ void k_adj(const long long* __restrict__ ei, unsigned* __restrict__ adj) {
  int i = blockIdx.x * blockDim.x + threadIdx.x;
  if (i < EE) {
    int s = (int)ei[i];
    int d = (int)ei[EE + i];
    atomicOr(&adj[s * (DD) + (d >> 5)], 1u << (d & 31));
  }
  if (i < NN) {
    atomicOr(&adj[i * (DD) + (i >> 5)], 1u << (i & 31));
  }
}

// ---------------------------------------------------------------------------
// Kernel 2: At[n][k] = (f16) A[k][n]   (so B fragments of g = h@A are contiguous)
// ---------------------------------------------------------------------------
__global__ void k_At(const float* __restrict__ A, _Float16* __restrict__ At) {
  int idx = blockIdx.x * 256 + threadIdx.x;
  int k = idx >> 8;
  int n = idx & 255;
  At[n * DD + k] = (_Float16)A[k * DD + n];
}

// ---------------------------------------------------------------------------
// Kernel 3: hb = f16( x @ W^T + b )   [8192 x 256]
// grid (64, 16): 128 rows / block (8 waves x 16 rows), 16 cols / block
// ---------------------------------------------------------------------------
__global__ __launch_bounds__(256) void k_gemm_h(const float* __restrict__ x,
                                                const float* __restrict__ W,
                                                const float* __restrict__ Wb,
                                                _Float16* __restrict__ hb) {
  const int lane = threadIdx.x & 31;
  const int wv   = threadIdx.x >> 5;
  const int sel  = lane >> 4;
  const int l16  = lane & 15;
  const int row0 = blockIdx.x * 128 + wv * 16;
  const int col0 = blockIdx.y * 16;

  v8f acc;
#pragma unroll
  for (int r = 0; r < 8; ++r) acc[r] = 0.0f;

#pragma unroll
  for (int kt = 0; kt < 8; ++kt) {
    const int k = kt * 32;
    v16h a = load_a_f32(x, DD, row0 + l16, k, sel);       // x rows
    v16h b = load_b_f32(W, DD, col0 + l16, k, sel);       // W[n][k] == (W^T)[k][n]
    acc = wmma16(a, b, acc);
  }
  const float bias = Wb[col0 + l16];
#pragma unroll
  for (int r = 0; r < 8; ++r) {
    const int row = row0 + r + 8 * sel;
    hb[row * DD + col0 + l16] = (_Float16)(acc[r] + bias);
  }
}

// ---------------------------------------------------------------------------
// Kernel 4: gb = f16( h @ A )   using hb (f16) and At (f16, transposed A)
// ---------------------------------------------------------------------------
__global__ __launch_bounds__(256) void k_gemm_g(const _Float16* __restrict__ hb,
                                                const _Float16* __restrict__ At,
                                                _Float16* __restrict__ gbm) {
  const int lane = threadIdx.x & 31;
  const int wv   = threadIdx.x >> 5;
  const int sel  = lane >> 4;
  const int l16  = lane & 15;
  const int row0 = blockIdx.x * 128 + wv * 16;
  const int col0 = blockIdx.y * 16;

  v8f acc;
#pragma unroll
  for (int r = 0; r < 8; ++r) acc[r] = 0.0f;

#pragma unroll
  for (int kt = 0; kt < 8; ++kt) {
    const int k = kt * 32;
    v16h a = load_a_h(hb, DD, row0 + l16, k, sel);
    v16h b = load_b_h(At, DD, col0 + l16, k, sel);
    acc = wmma16(a, b, acc);
  }
#pragma unroll
  for (int r = 0; r < 8; ++r) {
    const int row = row0 + r + 8 * sel;
    gbm[row * DD + col0 + l16] = (_Float16)acc[r];
  }
}

// ---------------------------------------------------------------------------
// Kernel 5: fused flash attention over the sparse-masked dense scores
//   S = G H^T + H G^T  (e + e^T), masked by adj bits, online softmax,
//   O += P * H, epilogue: h' = relu(O/l), sigmoid gate, blend with x.
// Column tiles of H and G are staged into LDS by the Tensor Data Mover
// (wave 0 issues the descriptors; TENSORcnt + block barrier publish them).
// grid 64 blocks x 256 threads; wave w owns rows [blk*128 + 16w, +16).
// ---------------------------------------------------------------------------
__global__ __launch_bounds__(256) void k_flash(const _Float16* __restrict__ hb,
                                               const _Float16* __restrict__ gbm,
                                               const unsigned* __restrict__ adj,
                                               const float* __restrict__ x,
                                               const float* __restrict__ gw,
                                               const float* __restrict__ gbia,
                                               float* __restrict__ out) {
  __shared__ _Float16 Hn[BN * DD];       // H col tile, [node][d]  (TDM) 16 KB
  __shared__ _Float16 Gn[BN * DD];       // G col tile, [node][d]  (TDM) 16 KB
  __shared__ _Float16 Ht[DD * BN];       // H col tile transposed        16 KB
  __shared__ _Float16 Pb[8][16 * BN];    // per-wave P staging            8 KB

  const int t    = threadIdx.x;
  const int lane = t & 31;
  const int wv   = t >> 5;
  const int sel  = lane >> 4;
  const int l16  = lane & 15;
  const int rowBase = blockIdx.x * 128 + wv * 16;
  const unsigned lds_Hn = (unsigned)(size_t)(&Hn[0]);   // LDS byte offsets
  const unsigned lds_Gn = (unsigned)(size_t)(&Gn[0]);

  // persistent accumulators: O covers 16 rows x 256 dims as 16 C-tiles
  v8f O[16];
#pragma unroll
  for (int i = 0; i < 16; ++i)
#pragma unroll
    for (int r = 0; r < 8; ++r) O[i][r] = 0.0f;

  float mrow[8], lrow[8];
#pragma unroll
  for (int r = 0; r < 8; ++r) { mrow[r] = -1e30f; lrow[r] = 0.0f; }

  // hoist row-side A fragments (G rows and H rows, all 8 K-tiles)
  v16h aG[8], aH[8];
#pragma unroll
  for (int kt = 0; kt < 8; ++kt) {
    aG[kt] = load_a_h(gbm, DD, rowBase + l16, kt * 32, sel);
    aH[kt] = load_a_h(hb,  DD, rowBase + l16, kt * 32, sel);
  }

  const int jstage = t >> 3;            // 0..31 : node within tile
  const int dstage = (t & 7) * 32;      // 0..224: dim group

  for (int it = 0; it < NN / BN; ++it) {
    __syncthreads();                     // previous iteration done with LDS
    {
      const int cb = it * BN;
      if (wv == 0) {                     // TDM: DMA both tiles into LDS
        tdm_load_1d(hb  + cb * DD, lds_Hn, (BN * DD * 2) / 8);
        tdm_load_1d(gbm + cb * DD, lds_Gn, (BN * DD * 2) / 8);
      }
      if (it + 1 < NN / BN) {            // prefetch next column tile
        __builtin_prefetch(hb  + (it + 1) * BN * DD + t * 8, 0, 0);
        __builtin_prefetch(gbm + (it + 1) * BN * DD + t * 8, 0, 0);
      }
      // transposed H copy built from registers (TDM cannot transpose)
#pragma unroll
      for (int c = 0; c < 32; c += 8) {
        v8h hv = *(const v8h*)(hb + (cb + jstage) * DD + dstage + c);
#pragma unroll
        for (int u = 0; u < 8; ++u)
          Ht[(dstage + c + u) * BN + jstage] = hv[u];
      }
      if (wv == 0) __builtin_amdgcn_s_wait_tensorcnt(0);   // drain TDM
    }
    __syncthreads();

    // ---- score tile S (16 rows x 32 cols) = G H^T + H G^T -----------------
    v8f S0, S1;
#pragma unroll
    for (int r = 0; r < 8; ++r) { S0[r] = 0.0f; S1[r] = 0.0f; }
#pragma unroll
    for (int kt = 0; kt < 8; ++kt) {
      const int k = kt * 32;
      v16h bH0 = load_b_h(Hn, DD, l16,      k, sel);
      v16h bH1 = load_b_h(Hn, DD, 16 + l16, k, sel);
      v16h bG0 = load_b_h(Gn, DD, l16,      k, sel);
      v16h bG1 = load_b_h(Gn, DD, 16 + l16, k, sel);
      S0 = wmma16(aG[kt], bH0, S0);
      S0 = wmma16(aH[kt], bG0, S0);
      S1 = wmma16(aG[kt], bH1, S1);
      S1 = wmma16(aH[kt], bG1, S1);
    }

    // ---- mask + online softmax, emit P (f16) to LDS -----------------------
#pragma unroll
    for (int r = 0; r < 8; ++r) {
      const int row = rowBase + r + 8 * sel;
      const unsigned w = adj[row * (DD) + it];
      const unsigned b0 = (w >> l16) & 1u;
      const unsigned b1 = (w >> (16 + l16)) & 1u;
      float s0 = b0 ? S0[r] : -1e30f;
      float s1 = b1 ? S1[r] : -1e30f;
      float tm = fmaxf(s0, s1);
      tm = fmaxf(tm, __shfl_xor(tm, 1, 16));
      tm = fmaxf(tm, __shfl_xor(tm, 2, 16));
      tm = fmaxf(tm, __shfl_xor(tm, 4, 16));
      tm = fmaxf(tm, __shfl_xor(tm, 8, 16));
      const float mnew = fmaxf(mrow[r], tm);
      const float scl  = __expf(mrow[r] - mnew);
      const float p0 = b0 ? __expf(s0 - mnew) : 0.0f;
      const float p1 = b1 ? __expf(s1 - mnew) : 0.0f;
      float ps = p0 + p1;
      ps += __shfl_xor(ps, 1, 16);
      ps += __shfl_xor(ps, 2, 16);
      ps += __shfl_xor(ps, 4, 16);
      ps += __shfl_xor(ps, 8, 16);
      lrow[r] = lrow[r] * scl + ps;
      mrow[r] = mnew;
#pragma unroll
      for (int dt = 0; dt < 16; ++dt) O[dt][r] *= scl;
      Pb[wv][(r + 8 * sel) * BN + l16]      = (_Float16)p0;
      Pb[wv][(r + 8 * sel) * BN + 16 + l16] = (_Float16)p1;
    }
    // same-wave LDS RAW: keep compiler + pipeline ordered
    asm volatile("s_wait_dscnt 0" ::: "memory");

    // ---- O += P(16x32) * H(32x256)  (B fragments from transposed tile) ----
    v16h aP;
    {
      const _Float16* p = &Pb[wv][l16 * BN + sel * 16];
      v8h lo = *(const v8h*)(p);
      v8h hi = *(const v8h*)(p + 8);
#pragma unroll
      for (int i = 0; i < 8; ++i) { aP[i] = lo[i]; aP[8 + i] = hi[i]; }
    }
#pragma unroll
    for (int dt = 0; dt < 16; ++dt) {
      v16h bT = *(const v16h*)(Ht + (dt * 16 + l16) * BN + sel * 16);
      O[dt] = wmma16(aP, bT, O[dt]);
    }
  }

  // ---- epilogue: normalize, relu, sigmoid gate, blend ---------------------
#pragma unroll
  for (int r = 0; r < 8; ++r) {
    const int row = rowBase + r + 8 * sel;
    const float inv = 1.0f / lrow[r];    // self-loop guarantees lrow > 0
    float part = 0.0f;
#pragma unroll
    for (int dt = 0; dt < 16; ++dt) {
      const int d = dt * 16 + l16;
      const float hp = fmaxf(O[dt][r] * inv, 0.0f);
      part += x[row * DD + d] * gw[d] + hp * gw[DD + d];
    }
    part += __shfl_xor(part, 1, 16);
    part += __shfl_xor(part, 2, 16);
    part += __shfl_xor(part, 4, 16);
    part += __shfl_xor(part, 8, 16);
    const float z = part + gbia[0];
    const float coeff = 1.0f / (1.0f + __expf(-z));
#pragma unroll
    for (int dt = 0; dt < 16; ++dt) {
      const int d = dt * 16 + l16;
      const float hp = fmaxf(O[dt][r] * inv, 0.0f);
      out[row * DD + d] = coeff * x[row * DD + d] + (1.0f - coeff) * hp;
    }
  }
}

// ---------------------------------------------------------------------------
// Launch
// ---------------------------------------------------------------------------
extern "C" void kernel_launch(void* const* d_in, const int* in_sizes, int n_in,
                              void* d_out, int out_size, void* d_ws, size_t ws_size,
                              hipStream_t stream) {
  const float*     x   = (const float*)d_in[0];
  const long long* ei  = (const long long*)d_in[1];
  const float*     Ww  = (const float*)d_in[2];
  const float*     Wb  = (const float*)d_in[3];
  const float*     A   = (const float*)d_in[4];
  const float*     gw  = (const float*)d_in[5];
  const float*     gbi = (const float*)d_in[6];
  float* out = (float*)d_out;

  char* ws = (char*)d_ws;
  unsigned*  adj = (unsigned*)(ws);                         // 8 MB bitmask
  _Float16*  hb  = (_Float16*)(ws + (8u  << 20));           // 4 MB f16 h
  _Float16*  gbm = (_Float16*)(ws + (12u << 20));           // 4 MB f16 g
  _Float16*  Atb = (_Float16*)(ws + (16u << 20));           // 128 KB f16 A^T

  k_zero<<<2048, 256, 0, stream>>>((uint4*)adj);            // 8MB / 16B / 256
  k_adj <<<512, 256, 0, stream>>>(ei, adj);
  k_At  <<<256, 256, 0, stream>>>(A, Atb);
  k_gemm_h<<<dim3(64, 16), 256, 0, stream>>>(x, Ww, Wb, hb);
  k_gemm_g<<<dim3(64, 16), 256, 0, stream>>>(hb, Atb, gbm);
  k_flash <<<64, 256, 0, stream>>>(hb, gbm, adj, x, gw, gbi, out);
}
